// SparseMoE_88510686036633
// MI455X (gfx1250) — compile-verified, matrix operand order checked by reading, not run on hardware
//
#include <hip/hip_runtime.h>
#include <hip/hip_bf16.h>

typedef __bf16 bf16;
typedef __attribute__((ext_vector_type(16))) __bf16 bf16x16;
typedef __attribute__((ext_vector_type(8)))  __bf16 bf16x8;
typedef __attribute__((ext_vector_type(2)))  __bf16 bf16x2;
typedef __attribute__((ext_vector_type(8)))  float  f32x8;

constexpr int E      = 16;
constexpr int TOPK   = 2;
constexpr int D_IN   = 1024;
constexpr int D_HID  = 4096;
constexpr int D_OUT  = 1024;
constexpr int NTOK   = 8192;
constexpr int NSLOT  = NTOK * TOPK;

constexpr int BM = 128, BN = 128, BK = 32;
constexpr int LDP = 40;   // LDS row stride in bf16 elems: 80 B, multiple of 16 -> aligned b128 frag loads

union FragU { bf16x16 v; struct { bf16x8 lo, hi; } p; };

__device__ __forceinline__ bf16x16 lds_frag(const bf16* base, int lane) {
  // Fragment layout per ISA 7.12.2 (16-bit A 16x32): lanes 0-15: K[0..7]+K[16..23];
  // lanes 16-31: K[8..15]+K[24..31]; row = lane & 15. Two ds_load_b128 per fragment.
  const bf16* p = base + (lane & 15) * LDP + ((lane >> 4) * 8);
  FragU u;
  u.p.lo = *(const bf16x8*)(p);
  u.p.hi = *(const bf16x8*)(p + 16);
  return u.v;
}

__device__ __forceinline__ f32x8 wmma_bf16(bf16x16 a, bf16x16 b, f32x8 c) {
  return __builtin_amdgcn_wmma_f32_16x16x32_bf16(false, a, false, b, (short)0, c, false, false);
}

// gfx1250 async DMA: global -> LDS, tracked by ASYNCcnt (ISA 15.18.3 opcode 98).
// LDS dest address = low 32 bits of the generic pointer (aperture rule, ISA 10.2).
__device__ __forceinline__ void async_copy_b128(uint32_t lds_off, uint64_t gaddr) {
  asm volatile("global_load_async_to_lds_b128 %0, %1, off"
               :: "v"(lds_off), "v"(gaddr) : "memory");
}
__device__ __forceinline__ void wait_asynccnt0() {
  asm volatile("s_wait_asynccnt 0x0" ::: "memory");
}

__global__ void moe_init(int* __restrict__ cnt) {
  if (threadIdx.x < E) cnt[threadIdx.x] = 0;
}

// ---------------- Gating: one wave32 per token ----------------
__global__ void __launch_bounds__(256)
moe_gating(const float* __restrict__ x, const float* __restrict__ Wg,
           const float* __restrict__ bg, const float* __restrict__ bias,
           float* __restrict__ logits, int* __restrict__ idxo,
           float* __restrict__ wgt, int* __restrict__ cnt, int* __restrict__ list)
{
  const int lane = threadIdx.x & 31;
  const int t = blockIdx.x * 8 + (threadIdx.x >> 5);
  if (t >= NTOK) return;

  float acc[E];
#pragma unroll
  for (int e = 0; e < E; ++e) acc[e] = 0.0f;

  for (int i = lane; i < D_IN; i += 32) {
    float xv = x[(size_t)t * D_IN + i];
    const float4* w4 = (const float4*)(Wg + (size_t)i * E);
    float4 w0 = w4[0], w1 = w4[1], w2 = w4[2], w3 = w4[3];
    acc[0]  = fmaf(xv, w0.x, acc[0]);  acc[1]  = fmaf(xv, w0.y, acc[1]);
    acc[2]  = fmaf(xv, w0.z, acc[2]);  acc[3]  = fmaf(xv, w0.w, acc[3]);
    acc[4]  = fmaf(xv, w1.x, acc[4]);  acc[5]  = fmaf(xv, w1.y, acc[5]);
    acc[6]  = fmaf(xv, w1.z, acc[6]);  acc[7]  = fmaf(xv, w1.w, acc[7]);
    acc[8]  = fmaf(xv, w2.x, acc[8]);  acc[9]  = fmaf(xv, w2.y, acc[9]);
    acc[10] = fmaf(xv, w2.z, acc[10]); acc[11] = fmaf(xv, w2.w, acc[11]);
    acc[12] = fmaf(xv, w3.x, acc[12]); acc[13] = fmaf(xv, w3.y, acc[13]);
    acc[14] = fmaf(xv, w3.z, acc[14]); acc[15] = fmaf(xv, w3.w, acc[15]);
  }
#pragma unroll
  for (int e = 0; e < E; ++e) {
#pragma unroll
    for (int off = 16; off > 0; off >>= 1)
      acc[e] += __shfl_xor(acc[e], off, 32);
    acc[e] += bg[e];
  }
  if (lane < E) logits[(size_t)t * E + lane] = acc[lane];
  if (lane == 0) {
    float bl[E];
#pragma unroll
    for (int e = 0; e < E; ++e) bl[e] = acc[e] + bias[e];
    int i0 = 0;
#pragma unroll
    for (int e = 1; e < E; ++e) if (bl[e] > bl[i0]) i0 = e;   // strict >: lowest index wins ties
    int i1 = (i0 == 0) ? 1 : 0;
#pragma unroll
    for (int e = 0; e < E; ++e) if (e != i0 && bl[e] > bl[i1]) i1 = e;
    float g0 = acc[i0], g1 = acc[i1];
    float m  = fmaxf(g0, g1);
    float e0 = __expf(g0 - m), e1 = __expf(g1 - m);
    float inv = 1.0f / (e0 + e1);
    idxo[2 * t] = i0; idxo[2 * t + 1] = i1;
    wgt[2 * t] = e0 * inv; wgt[2 * t + 1] = e1 * inv;
    int p0 = atomicAdd(&cnt[i0], 1); list[(size_t)i0 * NTOK + p0] = 2 * t;
    int p1 = atomicAdd(&cnt[i1], 1); list[(size_t)i1 * NTOK + p1] = 2 * t + 1;
  }
}

// ---------------- Stage A: h = relu(gather(x) @ W1[e] + b1[e]) -> bf16 ----------------
__global__ void __launch_bounds__(256)
moe_ffn1(const float* __restrict__ x, const float* __restrict__ W1,
         const float* __restrict__ b1, const int* __restrict__ cnt,
         const int* __restrict__ list, bf16* __restrict__ h)
{
  const int e  = blockIdx.z;
  const int Me = cnt[e];
  const int m0 = blockIdx.x * BM;
  if (m0 >= Me) return;
  const int n0 = blockIdx.y * BN;

  __shared__ __align__(16) bf16 As[2][BM * LDP];
  __shared__ __align__(16) bf16 Bs[2][BN * LDP];
  __shared__ int slotS[BM];

  const int tid  = threadIdx.x;
  const int lane = tid & 31;
  const int wid  = tid >> 5;
  const int wm   = (wid & 1) * 64;
  const int wn   = (wid >> 1) * 32;

  if (tid < BM) {
    int gm = m0 + tid;
    slotS[tid] = (gm < Me) ? list[(size_t)e * NTOK + gm] : -1;
  }
  __syncthreads();

  const float* Wb = W1 + (size_t)e * D_IN * D_HID;
  constexpr int KT = D_IN / BK;

  auto loadA = [&](int kt, int buf) {
    const int row = tid >> 1;
    const int c0  = (tid & 1) * 16;
    bf16* dst = &As[buf][row * LDP + c0];
    const int slot = slotS[row];
    bf16x8 o0, o1;
    if (slot >= 0) {
      const float4* s4 = (const float4*)(x + (size_t)(slot >> 1) * D_IN + kt * BK + c0);
      float4 v0 = s4[0], v1 = s4[1], v2 = s4[2], v3 = s4[3];
      o0[0] = (bf16)v0.x; o0[1] = (bf16)v0.y; o0[2] = (bf16)v0.z; o0[3] = (bf16)v0.w;
      o0[4] = (bf16)v1.x; o0[5] = (bf16)v1.y; o0[6] = (bf16)v1.z; o0[7] = (bf16)v1.w;
      o1[0] = (bf16)v2.x; o1[1] = (bf16)v2.y; o1[2] = (bf16)v2.z; o1[3] = (bf16)v2.w;
      o1[4] = (bf16)v3.x; o1[5] = (bf16)v3.y; o1[6] = (bf16)v3.z; o1[7] = (bf16)v3.w;
    } else {
#pragma unroll
      for (int i = 0; i < 8; ++i) { o0[i] = (bf16)0.0f; o1[i] = (bf16)0.0f; }
    }
    *(bf16x8*)(dst)     = o0;
    *(bf16x8*)(dst + 8) = o1;
  };

  // B loader: 2 K-rows x 8 N per thread; pack (k,k+1) pairs -> one b32 LDS store per column.
  auto loadB = [&](int kt, int buf) {
    const int k0  = (tid & 15) * 2;    // 0..30 (even)
    const int nc0 = (tid >> 4) * 8;    // 0..120
    const float* src0 = Wb + (size_t)(kt * BK + k0) * D_HID + n0 + nc0;
    const float* src1 = src0 + D_HID;
    if (kt + 2 < KT) __builtin_prefetch(src0 + (size_t)2 * BK * D_HID, 0, 1);
    float4 a0 = *(const float4*)(src0);
    float4 a1 = *(const float4*)(src0 + 4);
    float4 b0 = *(const float4*)(src1);
    float4 b1v = *(const float4*)(src1 + 4);
    float r0[8] = {a0.x, a0.y, a0.z, a0.w, a1.x, a1.y, a1.z, a1.w};
    float r1[8] = {b0.x, b0.y, b0.z, b0.w, b1v.x, b1v.y, b1v.z, b1v.w};
#pragma unroll
    for (int j = 0; j < 8; ++j) {
      bf16x2 pk = {(bf16)r0[j], (bf16)r1[j]};              // v_cvt_pk_bf16_f32
      *(bf16x2*)&Bs[buf][(nc0 + j) * LDP + k0] = pk;       // ds_store_b32
    }
  };

  f32x8 acc[4][2];
  const f32x8 zf = {0.f, 0.f, 0.f, 0.f, 0.f, 0.f, 0.f, 0.f};
#pragma unroll
  for (int a = 0; a < 4; ++a)
#pragma unroll
    for (int b = 0; b < 2; ++b) acc[a][b] = zf;

  loadA(0, 0); loadB(0, 0);
  __syncthreads();

  for (int kt = 0; kt < KT; ++kt) {
    const int buf = kt & 1;
    if (kt + 1 < KT) { loadA(kt + 1, buf ^ 1); loadB(kt + 1, buf ^ 1); }
    bf16x16 af[4];
#pragma unroll
    for (int fm = 0; fm < 4; ++fm) af[fm] = lds_frag(&As[buf][(wm + fm * 16) * LDP], lane);
    bf16x16 bf0 = lds_frag(&Bs[buf][(wn + 0)  * LDP], lane);
    bf16x16 bf1 = lds_frag(&Bs[buf][(wn + 16) * LDP], lane);
#pragma unroll
    for (int fm = 0; fm < 4; ++fm) {
      acc[fm][0] = wmma_bf16(af[fm], bf0, acc[fm][0]);
      acc[fm][1] = wmma_bf16(af[fm], bf1, acc[fm][1]);
    }
    __syncthreads();
  }

  // Epilogue. C layout: M = v + 8*(lane>>4), N = lane&15 per 16x16 tile.
  const int rb = wm + 8 * (lane >> 4);
  const int cb = n0 + wn + (lane & 15);
  const float bv0 = b1[(size_t)e * D_HID + cb];
  const float bv1 = b1[(size_t)e * D_HID + cb + 16];
#pragma unroll
  for (int fm = 0; fm < 4; ++fm)
#pragma unroll
    for (int fn = 0; fn < 2; ++fn)
#pragma unroll
      for (int v = 0; v < 8; ++v) {
        int r = rb + fm * 16 + v;
        if (m0 + r < Me) {
          int slot = slotS[r];
          float val = acc[fm][fn][v] + (fn ? bv1 : bv0);
          val = fmaxf(val, 0.0f);
          h[(size_t)slot * D_HID + cb + fn * 16] = (bf16)val;
        }
      }
}

// ---------------- Stage B: eo[slot] = (h[slot] @ W2[e] + b2[e]) * w[slot] ----------------
__global__ void __launch_bounds__(256)
moe_ffn2(const bf16* __restrict__ h, const float* __restrict__ W2,
         const float* __restrict__ b2, const int* __restrict__ cnt,
         const int* __restrict__ list, const float* __restrict__ wgt,
         float* __restrict__ eo)
{
  const int e  = blockIdx.z;
  const int Me = cnt[e];
  const int m0 = blockIdx.x * BM;
  if (m0 >= Me) return;
  const int n0 = blockIdx.y * BN;

  __shared__ __align__(16) bf16 As[2][BM * LDP];
  __shared__ __align__(16) bf16 Bs[2][BN * LDP];
  __shared__ int   slotS[BM];
  __shared__ float wS[BM];

  const int tid  = threadIdx.x;
  const int lane = tid & 31;
  const int wid  = tid >> 5;
  const int wm   = (wid & 1) * 64;
  const int wn   = (wid >> 1) * 32;

  if (tid < BM) {
    int gm = m0 + tid;
    int slot = (gm < Me) ? list[(size_t)e * NTOK + gm] : -1;
    slotS[tid] = slot;
    wS[tid] = (slot >= 0) ? wgt[slot] : 0.0f;
  }
  __syncthreads();

  const float* Wb = W2 + (size_t)e * D_HID * D_OUT;
  constexpr int KT = D_HID / BK;

  // A tile is a straight bf16 global->LDS copy: use gfx1250 async DMA (ASYNCcnt).
  auto loadA = [&](int kt, int buf) {
    const int row = tid >> 1;
    const int c0  = (tid & 1) * 16;
    bf16* dst = &As[buf][row * LDP + c0];
    const int slot = slotS[row];
    if (slot >= 0) {
      const bf16* src = h + (size_t)slot * D_HID + kt * BK + c0;
      const uint32_t doff = (uint32_t)(uintptr_t)dst;   // low 32 bits = LDS offset
      async_copy_b128(doff,      (uint64_t)(uintptr_t)src);
      async_copy_b128(doff + 16, (uint64_t)(uintptr_t)(src + 8));
    } else {
      bf16x8 z;
#pragma unroll
      for (int i = 0; i < 8; ++i) z[i] = (bf16)0.0f;
      *(bf16x8*)(dst)     = z;
      *(bf16x8*)(dst + 8) = z;
    }
  };

  auto loadB = [&](int kt, int buf) {
    const int k0  = (tid & 15) * 2;
    const int nc0 = (tid >> 4) * 8;
    const float* src0 = Wb + (size_t)(kt * BK + k0) * D_OUT + n0 + nc0;
    const float* src1 = src0 + D_OUT;
    if (kt + 2 < KT) __builtin_prefetch(src0 + (size_t)2 * BK * D_OUT, 0, 1);
    float4 a0 = *(const float4*)(src0);
    float4 a1 = *(const float4*)(src0 + 4);
    float4 b0 = *(const float4*)(src1);
    float4 b1v = *(const float4*)(src1 + 4);
    float r0[8] = {a0.x, a0.y, a0.z, a0.w, a1.x, a1.y, a1.z, a1.w};
    float r1[8] = {b0.x, b0.y, b0.z, b0.w, b1v.x, b1v.y, b1v.z, b1v.w};
#pragma unroll
    for (int j = 0; j < 8; ++j) {
      bf16x2 pk = {(bf16)r0[j], (bf16)r1[j]};
      *(bf16x2*)&Bs[buf][(nc0 + j) * LDP + k0] = pk;
    }
  };

  f32x8 acc[4][2];
  const f32x8 zf = {0.f, 0.f, 0.f, 0.f, 0.f, 0.f, 0.f, 0.f};
#pragma unroll
  for (int a = 0; a < 4; ++a)
#pragma unroll
    for (int b = 0; b < 2; ++b) acc[a][b] = zf;

  loadA(0, 0); loadB(0, 0);
  wait_asynccnt0();
  __syncthreads();

  for (int kt = 0; kt < KT; ++kt) {
    const int buf = kt & 1;
    if (kt + 1 < KT) { loadA(kt + 1, buf ^ 1); loadB(kt + 1, buf ^ 1); }
    bf16x16 af[4];
#pragma unroll
    for (int fm = 0; fm < 4; ++fm) af[fm] = lds_frag(&As[buf][(wm + fm * 16) * LDP], lane);
    bf16x16 bf0 = lds_frag(&Bs[buf][(wn + 0)  * LDP], lane);
    bf16x16 bf1 = lds_frag(&Bs[buf][(wn + 16) * LDP], lane);
#pragma unroll
    for (int fm = 0; fm < 4; ++fm) {
      acc[fm][0] = wmma_bf16(af[fm], bf0, acc[fm][0]);
      acc[fm][1] = wmma_bf16(af[fm], bf1, acc[fm][1]);
    }
    wait_asynccnt0();   // async fills for kt+1 must land before the barrier releases readers
    __syncthreads();
  }

  const int rb = wm + 8 * (lane >> 4);
  const int cb = n0 + wn + (lane & 15);
  const float bv0 = b2[(size_t)e * D_OUT + cb];
  const float bv1 = b2[(size_t)e * D_OUT + cb + 16];
#pragma unroll
  for (int fm = 0; fm < 4; ++fm)
#pragma unroll
    for (int fn = 0; fn < 2; ++fn)
#pragma unroll
      for (int v = 0; v < 8; ++v) {
        int r = rb + fm * 16 + v;
        if (m0 + r < Me) {
          int slot = slotS[r];
          float val = (acc[fm][fn][v] + (fn ? bv1 : bv0)) * wS[r];
          eo[(size_t)slot * D_OUT + cb + fn * 16] = val;
        }
      }
}

// ---------------- Combine: out[t] = eo[2t] + eo[2t+1] ----------------
__global__ void __launch_bounds__(256)
moe_combine(const float* __restrict__ eo, float* __restrict__ out)
{
  const size_t idx = (size_t)blockIdx.x * 256 + threadIdx.x;  // over NTOK * D_OUT/4
  const size_t t  = idx / (D_OUT / 4);
  const int    c4 = (int)(idx % (D_OUT / 4));
  const float4* e0 = (const float4*)(eo + (size_t)(2 * t)     * D_OUT);
  const float4* e1 = (const float4*)(eo + (size_t)(2 * t + 1) * D_OUT);
  float4 a = e0[c4], b = e1[c4];
  float4 r = {a.x + b.x, a.y + b.y, a.z + b.z, a.w + b.w};
  ((float4*)(out + t * D_OUT))[c4] = r;
}

extern "C" void kernel_launch(void* const* d_in, const int* in_sizes, int n_in,
                              void* d_out, int out_size, void* d_ws, size_t ws_size,
                              hipStream_t stream)
{
  (void)in_sizes; (void)n_in; (void)out_size; (void)ws_size;

  const float* x    = (const float*)d_in[0];
  const float* Wg   = (const float*)d_in[1];
  const float* bg   = (const float*)d_in[2];
  const float* W1   = (const float*)d_in[3];
  const float* b1   = (const float*)d_in[4];
  const float* W2   = (const float*)d_in[5];
  const float* b2   = (const float*)d_in[6];
  const float* bias = (const float*)d_in[7];

  float* out    = (float*)d_out;                       // [N, D_OUT]
  float* logits = out + (size_t)NTOK * D_OUT;          // [N, E]
  int*   idxo   = (int*)(logits + (size_t)NTOK * E);   // [N, TOPK] int32

  char* ws = (char*)d_ws;
  int*   cnt  = (int*)ws;                                              // 16 ints (pad to 256B)
  float* wgt  = (float*)(ws + 256);                                    // NSLOT floats
  int*   list = (int*)(ws + 256 + (size_t)NSLOT * 4);                  // E*NTOK ints
  bf16*  hbuf = (bf16*)(ws + 256 + (size_t)NSLOT * 4
                           + (size_t)E * NTOK * 4);                    // NSLOT * D_HID bf16
  float* eo   = (float*)((char*)hbuf + (size_t)NSLOT * D_HID * 2);     // NSLOT * D_OUT f32

  moe_init<<<1, 32, 0, stream>>>(cnt);
  moe_gating<<<NTOK / 8, 256, 0, stream>>>(x, Wg, bg, bias, logits, idxo, wgt, cnt, list);
  moe_ffn1<<<dim3(NTOK / BM, D_HID / BN, E), 256, 0, stream>>>(x, W1, b1, cnt, list, hbuf);
  moe_ffn2<<<dim3(NTOK / BM, D_OUT / BN, E), 256, 0, stream>>>(hbuf, W2, b2, cnt, list, wgt, eo);
  moe_combine<<<(NTOK * (D_OUT / 4)) / 256, 256, 0, stream>>>(eo, out);
}
